// MultiHeadDifferentialAttention_11562051960999
// MI455X (gfx1250) — matrix operand order repaired and checked by reading, not compiled, not run on hardware
//
#include <hip/hip_runtime.h>

// ---------------- problem constants ----------------
#define Bb   2
#define Nn   2048
#define Cc   768
#define Hh   12
#define TWO_C (2*Cc)            // 1536
#define Mrows (Bb*Nn)           // 4096
static __device__ constexpr float SCALE_Q   = 0.125f;   // D^-0.5 = 1/8
static __device__ constexpr float OUT_SCALE = 0.8f;     // 1 - (0.8 - 0.6*e^0)
static __device__ constexpr float LN_EPS    = 1e-5f;

typedef __bf16 bf16_t;
typedef __attribute__((ext_vector_type(16))) __bf16 v16bf;
typedef __attribute__((ext_vector_type(8)))  __bf16 v8bf;
typedef __attribute__((ext_vector_type(4)))  __bf16 v4bf;
typedef __attribute__((ext_vector_type(8)))  float  v8f;

// ---- gfx1250 async global->LDS copy (ASYNCcnt-tracked), with fallback ----
#if defined(__HIP_DEVICE_COMPILE__) && \
    __has_builtin(__builtin_amdgcn_global_load_async_to_lds_b128) && \
    __has_builtin(__builtin_amdgcn_s_wait_asynccnt)
#define HAVE_ASYNC_LDS 1
#else
#define HAVE_ASYNC_LDS 0
#endif

#if HAVE_ASYNC_LDS
typedef int v4i_raw __attribute__((vector_size(4 * sizeof(int))));
typedef __attribute__((address_space(1))) v4i_raw as1_v4i;   // global
typedef __attribute__((address_space(3))) v4i_raw as3_v4i;   // LDS
#endif

__device__ __forceinline__ void copy16_async(const bf16_t* g, bf16_t* l) {
#if HAVE_ASYNC_LDS
    __builtin_amdgcn_global_load_async_to_lds_b128(
        (as1_v4i*)g, (as3_v4i*)l, 0, 0);
#else
    *(uint4*)l = *(const uint4*)g;
#endif
}
__device__ __forceinline__ void wait_async() {
#if HAVE_ASYNC_LDS
    __builtin_amdgcn_s_wait_asynccnt(0);
#endif
}

// ---- CDNA5 wave32 WMMA fragment helpers (ISA 7.12.2) ----
// A (16x32, 16-bit): lane row=lane&15, khalf=lane>>4; per lane the 16 elements
// are one contiguous run: k in [khalf*16, +16).
__device__ __forceinline__ v16bf load_afrag(const bf16_t* row /*k=0 of row*/, int khalf) {
    return *(const v16bf*)(row + khalf * 16);
}
// B (32x16, 16-bit): lane col=lane&15, khalf; element e -> k = khalf*16+e.
// With a reduction-major layout (32 contiguous k per column): one 32B load.
__device__ __forceinline__ v16bf load_bfrag(const bf16_t* colk0 /*32B-aligned*/, int khalf) {
    return *(const v16bf*)(colk0 + khalf * 16);
}
// C/D (16x16 f32): lane col=lane&15; reg r -> row = r + 8*(lane>>4)

__device__ __forceinline__ float redmax16(float v) {
    #pragma unroll
    for (int o = 8; o > 0; o >>= 1) v = fmaxf(v, __shfl_xor(v, o, 32));
    return v;  // independent within each 16-lane half (xor < 16)
}
__device__ __forceinline__ float redsum16(float v) {
    #pragma unroll
    for (int o = 8; o > 0; o >>= 1) v += __shfl_xor(v, o, 32);
    return v;
}

// =====================================================================
// Kernel 1: C_bf16 = A_f32[M,K] @ B_f32[K,N]  (QKV projections)
// TR=false: row-major out.  TR=true: transposed out C^T[col*M + row] (for V).
// Block 256 thr (8 waves), tile 128x64, K-step 32. B tile k-major per column
// in LDS so B-fragments are single 32-byte ds loads.
// =====================================================================
#define GM 128
#define GN 64
#define GK 32
template<bool TR>
__global__ __launch_bounds__(256) void gemm_xw_bf16(
    const float* __restrict__ A, const float* __restrict__ Bm,
    bf16_t* __restrict__ Cout, int M, int K, int N)
{
    __shared__ __align__(32) bf16_t As[GM * GK];    // row-major, 8 KB
    __shared__ __align__(32) bf16_t BsT[GN * GK];   // col-major (k fastest), 4 KB
    const int tid = threadIdx.x, wave = tid >> 5, lane = tid & 31;
    const int lane15 = lane & 15, khalf = lane >> 4;
    const int rowBase = blockIdx.x * GM, colBase = blockIdx.y * GN;

    v8f acc[4] = {};
    for (int k0 = 0; k0 < K; k0 += GK) {
        #pragma unroll
        for (int i = 0; i < (GM * GK) / (256 * 4); ++i) {
            int ch = tid + 256 * i, r = ch >> 3, c4 = (ch & 7) * 4;
            float4 f = *(const float4*)&A[(size_t)(rowBase + r) * K + k0 + c4];
            v4bf h; h[0] = (bf16_t)f.x; h[1] = (bf16_t)f.y; h[2] = (bf16_t)f.z; h[3] = (bf16_t)f.w;
            *(v4bf*)&As[r * GK + c4] = h;
        }
        #pragma unroll
        for (int i = 0; i < (GN * GK) / (256 * 4); ++i) {
            int ch = tid + 256 * i, col = ch & 63, kq = ch >> 6;
            v4bf h;
            #pragma unroll
            for (int t = 0; t < 4; ++t)
                h[t] = (bf16_t)Bm[(size_t)(k0 + kq * 4 + t) * N + colBase + col];
            *(v4bf*)&BsT[col * GK + kq * 4] = h;
        }
        __syncthreads();
        v16bf af = load_afrag(&As[(wave * 16 + lane15) * GK], khalf);
        #pragma unroll
        for (int j = 0; j < 4; ++j) {
            v16bf bf_ = load_bfrag(&BsT[(j * 16 + lane15) * GK], khalf);
            acc[j] = __builtin_amdgcn_wmma_f32_16x16x32_bf16(
                false, af, false, bf_, (short)0, acc[j], false, false);
        }
        __syncthreads();
    }
    if (TR) {
        #pragma unroll
        for (int j = 0; j < 4; ++j) {
            v8bf o;
            #pragma unroll
            for (int r = 0; r < 8; ++r) o[r] = (bf16_t)acc[j][r];
            int col  = colBase + j * 16 + lane15;
            int row0 = rowBase + wave * 16 + 8 * khalf;   // 8 consecutive rows
            *(v8bf*)&Cout[(size_t)col * M + row0] = o;    // 16B store
        }
    } else {
        #pragma unroll
        for (int j = 0; j < 4; ++j)
            #pragma unroll
            for (int r = 0; r < 8; ++r) {
                int row = rowBase + wave * 16 + r + 8 * khalf;
                int col = colBase + j * 16 + lane15;
                Cout[(size_t)row * N + col] = (bf16_t)acc[j][r];
            }
    }
}

// =====================================================================
// Kernel 3: C_f32 = A_bf16[M,K] @ B_f32[K,N] + bias   (out projection)
// =====================================================================
__global__ __launch_bounds__(256) void gemm_proj(
    const bf16_t* __restrict__ A, const float* __restrict__ Bm,
    const float* __restrict__ bias, float* __restrict__ Cout,
    int M, int K, int N)
{
    __shared__ __align__(32) bf16_t As[GM * GK];
    __shared__ __align__(32) bf16_t BsT[GN * GK];
    const int tid = threadIdx.x, wave = tid >> 5, lane = tid & 31;
    const int lane15 = lane & 15, khalf = lane >> 4;
    const int rowBase = blockIdx.x * GM, colBase = blockIdx.y * GN;

    v8f acc[4] = {};
    for (int k0 = 0; k0 < K; k0 += GK) {
        #pragma unroll
        for (int i = 0; i < (GM * GK) / (256 * 8); ++i) {
            int ch = tid + 256 * i, r = ch >> 2, c8 = (ch & 3) * 8;
            copy16_async(&A[(size_t)(rowBase + r) * K + k0 + c8], &As[r * GK + c8]);
        }
        #pragma unroll
        for (int i = 0; i < (GN * GK) / (256 * 4); ++i) {
            int ch = tid + 256 * i, col = ch & 63, kq = ch >> 6;
            v4bf h;
            #pragma unroll
            for (int t = 0; t < 4; ++t)
                h[t] = (bf16_t)Bm[(size_t)(k0 + kq * 4 + t) * N + colBase + col];
            *(v4bf*)&BsT[col * GK + kq * 4] = h;
        }
        wait_async();
        __syncthreads();
        v16bf af = load_afrag(&As[(wave * 16 + lane15) * GK], khalf);
        #pragma unroll
        for (int j = 0; j < 4; ++j) {
            v16bf bf_ = load_bfrag(&BsT[(j * 16 + lane15) * GK], khalf);
            acc[j] = __builtin_amdgcn_wmma_f32_16x16x32_bf16(
                false, af, false, bf_, (short)0, acc[j], false, false);
        }
        __syncthreads();
    }
    #pragma unroll
    for (int j = 0; j < 4; ++j)
        #pragma unroll
        for (int r = 0; r < 8; ++r) {
            int row = rowBase + wave * 16 + r + 8 * khalf;
            int col = colBase + j * 16 + lane15;
            Cout[(size_t)row * N + col] = acc[j][r] + bias[col];
        }
}

// =====================================================================
// Kernel 2: differential flash attention + fused LayerNorm epilogue.
// Grid (B*H, N/128); block 256 thr = 8 waves, wave owns 16 query rows.
// K tile key-major (score B-frags contiguous along d); V pre-transposed in
// global (VT[h*128+d][b*N+n]) so P@V B-frags are contiguous along keys.
// K/V tiles are double-buffered and fetched with ASYNCcnt-tracked
// global_load_async_to_lds_b128, overlapping HBM/L2 latency with WMMAs.
// =====================================================================
#define AT_WAVES 8
__global__ __launch_bounds__(32 * AT_WAVES) void diff_attn(
    const bf16_t* __restrict__ Qg, const bf16_t* __restrict__ Kg,
    const bf16_t* __restrict__ VTg, const float* __restrict__ lambda_p,
    const float* __restrict__ ln_w, const float* __restrict__ ln_b,
    bf16_t* __restrict__ Y)
{
    __shared__ __align__(32) bf16_t Kt [2][32 * 128];       // [key][d]    16 KB
    __shared__ __align__(32) bf16_t VtT[2][128 * 32];       // [vdim][key] 16 KB
    __shared__ __align__(32) bf16_t Ps[AT_WAVES][16 * 32];  // P relayout   8 KB

    const int bh = blockIdx.x, b = bh / Hh, h = bh % Hh;
    const int qBase = blockIdx.y * (16 * AT_WAVES);
    const int tid = threadIdx.x, wave = tid >> 5, lane = tid & 31;
    const int lane15 = lane & 15, khalf = lane >> 4;
    const int myQ = qBase + wave * 16;
    const int NT = 32 * AT_WAVES;   // threads per block

    const bf16_t* Qh  = Qg  + (size_t)b * Nn * TWO_C + h * 128;
    const bf16_t* Kh  = Kg  + (size_t)b * Nn * TWO_C + h * 128;
    const bf16_t* VTh = VTg + (size_t)(h * 128) * Mrows + (size_t)b * Nn;

    // Q fragments (one contiguous 16-element run per lane), SCALE folded in
    v16bf q1f[2], q2f[2];
    {
        const bf16_t* qp = Qh + (size_t)(myQ + lane15) * TWO_C;
        #pragma unroll
        for (int half = 0; half < 2; ++half) {
            q1f[half] = load_afrag(qp + half * 32, khalf);
            q2f[half] = load_afrag(qp + 64 + half * 32, khalf);
        }
        #pragma unroll
        for (int e = 0; e < 16; ++e) {
            q1f[0][e] = (bf16_t)((float)q1f[0][e] * SCALE_Q);
            q1f[1][e] = (bf16_t)((float)q1f[1][e] * SCALE_Q);
            q2f[0][e] = (bf16_t)((float)q2f[0][e] * SCALE_Q);
            q2f[1][e] = (bf16_t)((float)q2f[1][e] * SCALE_Q);
        }
    }

    // issue async loads of one 32-key tile into buffer `buf`
    auto issue_tile = [&](int kt, int buf) {
        for (int i = tid; i < (32 * 128) / 8; i += NT) {
            int r = i >> 4, c = (i & 15) * 8;
            copy16_async(&Kh[(size_t)(kt + r) * TWO_C + c], &Kt[buf][r * 128 + c]);
        }
        for (int i = tid; i < (128 * 32) / 8; i += NT) {
            int d = i >> 2, c = (i & 3) * 8;
            copy16_async(&VTh[(size_t)d * Mrows + kt + c], &VtT[buf][d * 32 + c]);
        }
    };

    v8f O1[8] = {}, O2[8] = {};
    float m1[8], l1[8], m2[8], l2[8];
    #pragma unroll
    for (int r = 0; r < 8; ++r) { m1[r] = -1e30f; l1[r] = 0.f; m2[r] = -1e30f; l2[r] = 0.f; }

    issue_tile(0, 0);
    int cur = 0;
    for (int kt = 0; kt < Nn; kt += 32) {
        wait_async();          // my async writes into Kt[cur]/VtT[cur] done
        __syncthreads();       // everyone's done -> tile valid; prev reads done
        if (kt + 32 < Nn) issue_tile(kt + 32, cur ^ 1);

        // scores: S1,S2 are 16x32 (two C-frags each); K-dim 64 = 2 WMMAs
        v8f s1[2], s2[2];
        #pragma unroll
        for (int j = 0; j < 2; ++j) {
            const bf16_t* kp = &Kt[cur][(j * 16 + lane15) * 128];
            v8f t = {};
            t = __builtin_amdgcn_wmma_f32_16x16x32_bf16(false, q1f[0], false, load_bfrag(kp +  0, khalf), (short)0, t, false, false);
            t = __builtin_amdgcn_wmma_f32_16x16x32_bf16(false, q1f[1], false, load_bfrag(kp + 32, khalf), (short)0, t, false, false);
            s1[j] = t;
            v8f u = {};
            u = __builtin_amdgcn_wmma_f32_16x16x32_bf16(false, q2f[0], false, load_bfrag(kp + 64, khalf), (short)0, u, false, false);
            u = __builtin_amdgcn_wmma_f32_16x16x32_bf16(false, q2f[1], false, load_bfrag(kp + 96, khalf), (short)0, u, false, false);
            s2[j] = u;
        }

        // two online-softmax streams sharing the V tile
        #pragma unroll
        for (int st = 0; st < 2; ++st) {
            v8f*   s = (st == 0) ? s1 : s2;
            float* m = (st == 0) ? m1 : m2;
            float* l = (st == 0) ? l1 : l2;
            v8f*   O = (st == 0) ? O1 : O2;

            float alpha[8];
            #pragma unroll
            for (int r = 0; r < 8; ++r) {
                float mx = redmax16(fmaxf(s[0][r], s[1][r]));
                float mnew = fmaxf(m[r], mx);
                alpha[r] = __expf(m[r] - mnew);
                m[r] = mnew;
                float p0 = __expf(s[0][r] - mnew);
                float p1 = __expf(s[1][r] - mnew);
                s[0][r] = p0; s[1][r] = p1;
                l[r] = l[r] * alpha[r] + redsum16(p0 + p1);
            }
            // C-layout -> LDS row-major 16x32 -> A-fragment (2 x b128 reads)
            #pragma unroll
            for (int j = 0; j < 2; ++j)
                #pragma unroll
                for (int r = 0; r < 8; ++r)
                    Ps[wave][(r + 8 * khalf) * 32 + j * 16 + lane15] = (bf16_t)s[j][r];
            v16bf pa = load_afrag(&Ps[wave][lane15 * 32], khalf);
            // O = O*alpha + P @ V
            #pragma unroll
            for (int j = 0; j < 8; ++j) {
                v8f o = O[j];
                #pragma unroll
                for (int r = 0; r < 8; ++r) o[r] *= alpha[r];
                v16bf vb = load_bfrag(&VtT[cur][(j * 16 + lane15) * 32], khalf);
                O[j] = __builtin_amdgcn_wmma_f32_16x16x32_bf16(
                    false, pa, false, vb, (short)0, o, false, false);
            }
        }
        cur ^= 1;
    }

    // epilogue: combine streams, LayerNorm over 128 dims, scale, store bf16
    const float lam = lambda_p[0];
    float lw[8], lb[8];
    #pragma unroll
    for (int j = 0; j < 8; ++j) { lw[j] = ln_w[j * 16 + lane15]; lb[j] = ln_b[j * 16 + lane15]; }
    #pragma unroll
    for (int r = 0; r < 8; ++r) {
        const float i1 = 1.f / l1[r], i2 = lam / l2[r];
        float xv[8], su = 0.f, sq = 0.f;
        #pragma unroll
        for (int j = 0; j < 8; ++j) {
            float xx = O1[j][r] * i1 - O2[j][r] * i2;
            xv[j] = xx; su += xx; sq += xx * xx;
        }
        su = redsum16(su); sq = redsum16(sq);
        const float mean = su * (1.f / 128.f);
        const float var  = sq * (1.f / 128.f) - mean * mean;
        const float rstd = rsqrtf(var + LN_EPS);
        const int row = myQ + r + 8 * khalf;
        const size_t base = ((size_t)b * Nn + row) * TWO_C + h * 128;
        #pragma unroll
        for (int j = 0; j < 8; ++j)
            Y[base + j * 16 + lane15] =
                (bf16_t)(((xv[j] - mean) * rstd * lw[j] + lb[j]) * OUT_SCALE);
    }
}

// =====================================================================
extern "C" void kernel_launch(void* const* d_in, const int* in_sizes, int n_in,
                              void* d_out, int out_size, void* d_ws, size_t ws_size,
                              hipStream_t stream) {
    const float* x      = (const float*)d_in[0];
    const float* wq     = (const float*)d_in[1];
    const float* wk     = (const float*)d_in[2];
    const float* wv     = (const float*)d_in[3];
    const float* lambda = (const float*)d_in[4];
    const float* ln_w   = (const float*)d_in[5];
    const float* ln_b   = (const float*)d_in[6];
    const float* proj_w = (const float*)d_in[7];
    const float* proj_b = (const float*)d_in[8];
    float* out = (float*)d_out;

    // workspace: Q,K row-major, VT transposed, Y; each 4096x1536 bf16 (~50 MB)
    const size_t SZ = (size_t)Mrows * TWO_C;
    bf16_t* Qb  = (bf16_t*)d_ws;
    bf16_t* Kb  = Qb + SZ;
    bf16_t* VTb = Kb + SZ;
    bf16_t* Yb  = VTb + SZ;

    dim3 g1(Mrows / GM, TWO_C / GN);
    gemm_xw_bf16<false><<<g1, 256, 0, stream>>>(x, wq, Qb,  Mrows, Cc, TWO_C);
    gemm_xw_bf16<false><<<g1, 256, 0, stream>>>(x, wk, Kb,  Mrows, Cc, TWO_C);
    gemm_xw_bf16<true ><<<g1, 256, 0, stream>>>(x, wv, VTb, Mrows, Cc, TWO_C);

    dim3 g2(Bb * Hh, Nn / (16 * AT_WAVES));
    diff_attn<<<g2, 32 * AT_WAVES, 0, stream>>>(Qb, Kb, VTb, lambda, ln_w, ln_b, Yb);

    dim3 g3(Mrows / GM, Cc / GN);
    gemm_proj<<<g3, 256, 0, stream>>>(Yb, proj_w, proj_b, out, Mrows, TWO_C, Cc);
}
